// DGLTemporalGAT_23922967839178
// MI455X (gfx1250) — compile-verified, matrix operand order checked by reading, not run on hardware
//
#include <hip/hip_runtime.h>

// Problem constants
#define BB 32
#define WW 512
#define FF 256
#define HH 8
#define DD 32
#define GM (BB*WW)   // 16384 GEMM rows
#define GN 256
#define GK 256

typedef __attribute__((ext_vector_type(16))) _Float16 v16h;
typedef __attribute__((ext_vector_type(8)))  float    v8f;
typedef __attribute__((ext_vector_type(4)))  float    v4f;
typedef __attribute__((ext_vector_type(4)))  int      v4i;

union AB16 { v4i q[2]; v16h h16; _Float16 h[16]; };
union Q8   { v4i q;    _Float16 h[8]; };

__device__ __forceinline__ float lrelu(float v) { return v > 0.f ? v : 0.2f * v; }

// ---------------------------------------------------------------------------
// Weight convert + transpose: WT[n][k] = (f16) W[k][n]   (256x256)
// ---------------------------------------------------------------------------
__global__ __launch_bounds__(256)
void k_w_to_h_t(const float* __restrict__ Win, _Float16* __restrict__ WT) {
    int i = blockIdx.x * blockDim.x + threadIdx.x;     // 65536
    int r = i >> 8, c = i & 255;
    WT[(size_t)c * 256 + r] = (_Float16)Win[i];
}

// ---------------------------------------------------------------------------
// WMMA GEMM: C[M x 256] = f16(A[M x 256]) * BT^T + bias
//   A f32 row-major (converted to f16 while staging to LDS)
//   BT f16 [N][K] (pre-transposed) so fragments read contiguous K runs
//   Workgroup: 256 threads = 8 waves; 64x64 tile; wave tile 16x32.
// ---------------------------------------------------------------------------
__global__ __launch_bounds__(256)
void k_gemm(const float* __restrict__ A, const _Float16* __restrict__ BT,
            const float* __restrict__ bias, float* __restrict__ C) {
    __shared__ __align__(16) _Float16 lA[64 * 40];   // padded stride 40 halves (80B)
    __shared__ __align__(16) _Float16 lB[64 * 40];

    const int tid  = threadIdx.x;
    const int wv   = tid >> 5, lane = tid & 31;
    const int tl   = lane & 15, sel = lane >> 4;     // sel: K-half selector (ISA 16-bit layout)
    const int m0g  = blockIdx.x * 64, n0g = blockIdx.y * 64;
    const int mw   = (wv >> 1) * 16, nw = (wv & 1) * 32;
    const int lr   = tid >> 2, lc = (tid & 3) * 8;   // 64 rows x 32 cols staged by 256 thr

    v8f acc0 = {}, acc1 = {};

    for (int kt = 0; kt < GK / 32; ++kt) {
        const int k0 = kt * 32;
        v4f f0 = *(const v4f*)(A + (size_t)(m0g + lr) * GK + k0 + lc);
        v4f f1 = *(const v4f*)(A + (size_t)(m0g + lr) * GK + k0 + lc + 4);
        v4i bq = *(const v4i*)(BT + (size_t)(n0g + lr) * GK + k0 + lc);
        if (kt + 1 < GK / 32)
            __builtin_prefetch(BT + (size_t)(n0g + lr) * GK + k0 + 32 + lc, 0, 1);

        __syncthreads();
        Q8 qa;
        #pragma unroll
        for (int i = 0; i < 4; ++i) { qa.h[i] = (_Float16)f0[i]; qa.h[4 + i] = (_Float16)f1[i]; }
        *(v4i*)(lA + lr * 40 + lc) = qa.q;
        *(v4i*)(lB + lr * 40 + lc) = bq;
        __syncthreads();

        // A fragment: row = mw+tl; lanes<16 hold K {0..7,16..23}, lanes>=16 K {8..15,24..31}
        AB16 af, b0, b1;
        const _Float16* ap  = lA + (mw + tl) * 40 + sel * 8;
        af.q[0] = *(const v4i*)(ap);
        af.q[1] = *(const v4i*)(ap + 16);
        const _Float16* bp0 = lB + (nw + tl) * 40 + sel * 8;
        b0.q[0] = *(const v4i*)(bp0);
        b0.q[1] = *(const v4i*)(bp0 + 16);
        const _Float16* bp1 = lB + (nw + 16 + tl) * 40 + sel * 8;
        b1.q[0] = *(const v4i*)(bp1);
        b1.q[1] = *(const v4i*)(bp1 + 16);

        acc0 = __builtin_amdgcn_wmma_f32_16x16x32_f16(false, af.h16, false, b0.h16,
                                                      (short)0, acc0, false, false);
        acc1 = __builtin_amdgcn_wmma_f32_16x16x32_f16(false, af.h16, false, b1.h16,
                                                      (short)0, acc1, false, false);
    }

    // C/D layout: lane holds N = lane&15; VGPR v -> M = v + 8*(lane>=16)
    const int   n0  = n0g + nw + tl;
    const float bv0 = bias ? bias[n0] : 0.f;
    const float bv1 = bias ? bias[n0 + 16] : 0.f;
    #pragma unroll
    for (int v = 0; v < 8; ++v) {
        const int mrow = m0g + mw + v + sel * 8;
        float* cp = C + (size_t)mrow * GN;
        cp[n0]      = acc0[v] + bv0;
        cp[n0 + 16] = acc1[v] + bv1;
    }
}

// ---------------------------------------------------------------------------
// el/er attention scores: elT/erT layout [b,h,s] (s contiguous)
// ---------------------------------------------------------------------------
__global__ __launch_bounds__(256)
void k_scores(const float* __restrict__ z, const float* __restrict__ al,
              const float* __restrict__ ar, float* __restrict__ elT,
              float* __restrict__ erT) {
    int i = blockIdx.x * blockDim.x + threadIdx.x;   // B*W*H = 131072
    int h = i & 7, w = (i >> 3) & 511, b = i >> 12;
    const float* zr  = z + ((size_t)(b * WW + w) * GN + h * DD);
    const float* alp = al + h * DD;
    const float* arp = ar + h * DD;
    float sl = 0.f, sr = 0.f;
    #pragma unroll
    for (int d = 0; d < DD; ++d) { float zv = zr[d]; sl += zv * alp[d]; sr += zv * arp[d]; }
    size_t o = (size_t)(b * HH + h) * WW + w;
    elT[o] = sl; erT[o] = sr;
}

// zt[b,h,d,s] = (f16) z[b,s,h,d]   (s contiguous for WMMA B fragments)
__global__ __launch_bounds__(256)
void k_zt(const float* __restrict__ z, _Float16* __restrict__ zt) {
    int i = blockIdx.x * blockDim.x + threadIdx.x;   // 4M
    int s = i & 511, d = (i >> 9) & 31, h = (i >> 14) & 7, b = i >> 17;
    zt[i] = (_Float16)z[((size_t)(b * WW + s) * GN) + h * DD + d];
}

// per-(b,h) max over s of el  (256 rows, one thread each)
__global__ __launch_bounds__(256)
void k_rowmax(const float* __restrict__ elT, float* __restrict__ m0) {
    int bh = threadIdx.x;
    const float* p = elT + (size_t)bh * WW;
    float m = -3.4e38f;
    for (int s = 0; s < WW; ++s) m = fmaxf(m, p[s]);
    m0[bh] = m;
}

// ---------------------------------------------------------------------------
// Attention: per wave a 16(t) x 32(d) tile for one (b,h).
// Softmax numerators computed once and reused as A fragment AND denominator.
// ---------------------------------------------------------------------------
__global__ __launch_bounds__(128)
void k_attn(const float* __restrict__ elT, const float* __restrict__ erT,
            const float* __restrict__ m0, const _Float16* __restrict__ zt,
            const float* __restrict__ gbias, float* __restrict__ outF) {
    __shared__ float ps[4][32];
    const int wv = threadIdx.x >> 5, lane = threadIdx.x & 31;
    const int tl = lane & 15, sel = lane >> 4;
    const int b = blockIdx.z, h = blockIdx.y;
    const int tbase = blockIdx.x * 64 + wv * 16;
    const int bh = b * HH + h;
    const int t = tbase + tl;

    const float er_t = erT[(size_t)bh * WW + t];
    const float mrow = lrelu(m0[bh] + er_t);            // lrelu monotone -> true row max
    const float* el  = elT + (size_t)bh * WW;
    const _Float16* zb = zt + (size_t)bh * (DD * WW);

    v8f acc0 = {}, acc1 = {};
    float psum = 0.f;

    for (int s0 = 0; s0 < WW; s0 += 32) {
        const int off = s0 + sel * 8;                   // K-half per ISA A layout
        v4f e0 = *(const v4f*)(el + off);
        v4f e1 = *(const v4f*)(el + off + 4);
        v4f e2 = *(const v4f*)(el + off + 16);
        v4f e3 = *(const v4f*)(el + off + 20);
        AB16 af;
        #pragma unroll
        for (int i = 0; i < 4; ++i) {
            float p0 = __expf(lrelu(e0[i] + er_t) - mrow);
            float p1 = __expf(lrelu(e1[i] + er_t) - mrow);
            float p2 = __expf(lrelu(e2[i] + er_t) - mrow);
            float p3 = __expf(lrelu(e3[i] + er_t) - mrow);
            psum += p0 + p1 + p2 + p3;
            af.h[i]      = (_Float16)p0;   // K off+i
            af.h[4 + i]  = (_Float16)p1;   // K off+4+i
            af.h[8 + i]  = (_Float16)p2;   // K off+16+i
            af.h[12 + i] = (_Float16)p3;   // K off+20+i
        }

        AB16 b0f, b1f;
        const _Float16* z0 = zb + (size_t)tl * WW + off;          // d = tl
        b0f.q[0] = *(const v4i*)(z0);
        b0f.q[1] = *(const v4i*)(z0 + 16);
        const _Float16* z1 = zb + (size_t)(16 + tl) * WW + off;   // d = 16+tl
        b1f.q[0] = *(const v4i*)(z1);
        b1f.q[1] = *(const v4i*)(z1 + 16);

        acc0 = __builtin_amdgcn_wmma_f32_16x16x32_f16(false, af.h16, false, b0f.h16,
                                                      (short)0, acc0, false, false);
        acc1 = __builtin_amdgcn_wmma_f32_16x16x32_f16(false, af.h16, false, b1f.h16,
                                                      (short)0, acc1, false, false);
    }

    // lanes (tl) and (tl+16) together cover all s for row t -> denominator
    ps[wv][lane] = psum;
    __syncthreads();

    #pragma unroll
    for (int v = 0; v < 8; ++v) {
        const int trow = tbase + v + sel * 8;
        const float dnm = ps[wv][v + sel * 8] + ps[wv][v + sel * 8 + 16];
        const float inv = 1.f / dnm;
        const int d0 = tl, d1 = 16 + tl;
        float* op = outF + ((size_t)(b * WW + trow) * GN) + h * DD;
        op[d0] = acc0[v] * inv + gbias[h * DD + d0];
        op[d1] = acc1[v] * inv + gbias[h * DD + d1];
    }
}

// ---------------------------------------------------------------------------
extern "C" void kernel_launch(void* const* d_in, const int* in_sizes, int n_in,
                              void* d_out, int out_size, void* d_ws, size_t ws_size,
                              hipStream_t stream) {
    (void)in_sizes; (void)n_in; (void)out_size; (void)ws_size;
    const float* x      = (const float*)d_in[0];
    const float* Wfc    = (const float*)d_in[1];
    const float* attn_l = (const float*)d_in[2];
    const float* attn_r = (const float*)d_in[3];
    const float* gbias  = (const float*)d_in[4];
    const float* Wproj  = (const float*)d_in[5];
    const float* bproj  = (const float*)d_in[6];
    float* out = (float*)d_out;

    char* ws = (char*)d_ws;
    const size_t zOff   = 0;                                      // z f32: 16 MB
    const size_t ztOff  = zOff   + (size_t)GM * GN * 4;           // zt f16: 8 MB
    const size_t aoOff  = ztOff  + (size_t)BB * HH * DD * WW * 2; // attn out f32: 16 MB
    const size_t wfcOff = aoOff  + (size_t)GM * GN * 4;
    const size_t wprOff = wfcOff + (size_t)256 * 256 * 2;
    const size_t elOff  = wprOff + (size_t)256 * 256 * 2;
    const size_t erOff  = elOff  + (size_t)BB * HH * WW * 4;
    const size_t m0Off  = erOff  + (size_t)BB * HH * WW * 4;

    float*     z    = (float*)(ws + zOff);
    _Float16*  zt   = (_Float16*)(ws + ztOff);
    float*     ao   = (float*)(ws + aoOff);
    _Float16*  wfcT = (_Float16*)(ws + wfcOff);
    _Float16*  wprT = (_Float16*)(ws + wprOff);
    float*     elT  = (float*)(ws + elOff);
    float*     erT  = (float*)(ws + erOff);
    float*     m0   = (float*)(ws + m0Off);

    k_w_to_h_t<<<256, 256, 0, stream>>>(Wfc,   wfcT);
    k_w_to_h_t<<<256, 256, 0, stream>>>(Wproj, wprT);
    k_gemm<<<dim3(GM / 64, GN / 64), 256, 0, stream>>>(x, wfcT, nullptr, z);
    k_scores<<<(BB * WW * HH) / 256, 256, 0, stream>>>(z, attn_l, attn_r, elT, erT);
    k_zt<<<(BB * HH * DD * WW) / 256, 256, 0, stream>>>(z, zt);
    k_rowmax<<<1, 256, 0, stream>>>(elT, m0);
    k_attn<<<dim3(WW / 64, HH, BB), 128, 0, stream>>>(elT, erT, m0, zt, gbias, ao);
    k_gemm<<<dim3(GM / 64, GN / 64), 256, 0, stream>>>(ao, wprT, bproj, out);
}
